// Model_15590731285219
// MI455X (gfx1250) — matrix operand order repaired and checked by reading, not compiled, not run on hardware
//
#include <hip/hip_runtime.h>
#include <hip/hip_bf16.h>
#include <math.h>

// ---- problem constants (match reference) ----
#define B_    4
#define N_    128
#define H_    512
#define NH_   8
#define HD_   64
#define DEPTH_ 4
#define TN_   512          // B*N
#define MH_   2048         // 4*H
#define EHD_  64
#define NN_   (N_*N_)      // 16384
#define BNN_  (B_*N_*N_)   // 65536
#define SCALE_ 0.125f      // HD^-0.5

typedef __attribute__((ext_vector_type(16))) __bf16 v16bf;
typedef __attribute__((ext_vector_type(8)))  __bf16 v8bf;
typedef __attribute__((ext_vector_type(8)))  float  v8f;

template <int P> struct IC { static constexpr int v = P; };

// ---------------- helpers ----------------
__device__ __forceinline__ float gelu_f(float x) {
    const float c0 = 0.7978845608028654f; // sqrt(2/pi)
    float t = tanhf(c0 * (x + 0.044715f * x * x * x));
    return 0.5f * x * (1.0f + t);
}

__device__ __forceinline__ unsigned pack_bf16(float a, float b) {
    union { __bf16 h[2]; unsigned u; } p;
    p.h[0] = (__bf16)a; p.h[1] = (__bf16)b;
    return p.u;           // lowers to v_cvt_pk_bf16_f32
}

// A fragment (16x32, bf16) from LDS, row-major with leading dim ld (halves).
// ISA layout: lanes 0-15 hold M=lane, K = {0..7, 16..23}; lanes 16-31 K = {8..15, 24..31}.
__device__ __forceinline__ v16bf frag_a(const __bf16* Sm, int ld, int row0, int kbase, int lane) {
    int row = row0 + (lane & 15);
    int ko  = kbase + ((lane & 16) ? 8 : 0);
    const __bf16* p = Sm + row * ld + ko;
    v8bf lo = *(const v8bf*)(p);
    v8bf hi = *(const v8bf*)(p + 16);
    v16bf a;
#pragma unroll
    for (int i = 0; i < 8; ++i) { a[i] = lo[i]; a[i + 8] = hi[i]; }
    return a;
}

// B fragment (32x16, bf16) from LDS staged TRANSPOSED (N-major, ld halves per column).
// ISA layout: lanes 0-15 hold N=lane, K = {0..7, 16..23}; lanes 16-31 K = {8..15, 24..31}.
__device__ __forceinline__ v16bf frag_b(const __bf16* SmT, int ld, int col0, int kbase, int lane) {
    int col = col0 + (lane & 15);
    int ko  = kbase + ((lane & 16) ? 8 : 0);
    const __bf16* p = SmT + col * ld + ko;
    v8bf lo = *(const v8bf*)(p);
    v8bf hi = *(const v8bf*)(p + 16);
    v16bf b;
#pragma unroll
    for (int i = 0; i < 8; ++i) { b[i] = lo[i]; b[i + 8] = hi[i]; }
    return b;
}

__device__ __forceinline__ v8f wmma_bf16(v16bf a, v16bf b, v8f c) {
    return __builtin_amdgcn_wmma_f32_16x16x32_bf16(false, a, false, b, (short)0, c, false, false);
}

__device__ __forceinline__ void atomicMaxF(float* addr, float val) {
    int* ia = (int*)addr;
    int cur = __float_as_int(*addr);
    while (val > __int_as_float(cur)) {
        int old = atomicCAS(ia, cur, __float_as_int(val));
        if (old == cur) break;
        cur = old;
    }
}

// ---------------- weight convert + transpose (fp32 KxN -> bf16 NxK) ----------------
__global__ __launch_bounds__(256)
void convT_kernel(const float* __restrict__ W, __bf16* __restrict__ WT, int K, int Nc) {
    long long base = (long long)blockIdx.z * K * Nc;
    int idx = blockIdx.x * 256 + threadIdx.x;         // output-linear: coalesced writes
    if (idx >= K * Nc) return;
    int n = idx / K, k = idx - n * K;
    WT[base + idx] = (__bf16)W[base + (long long)k * Nc + n];
}

// ---------------- generic WMMA GEMM ----------------
// C[M,N] = act(A[M,K] @ B[K,N] + bias) (+ Res); B passed pre-transposed bf16 (N x K).
// Row-major, dims multiples of (128,64,32). 256 threads = 8 waves; tile 128x64.
template <int ACT, bool BIAS, bool RES>
__global__ __launch_bounds__(256)
void gemm_wmma_kernel(const float* __restrict__ A, const __bf16* __restrict__ BT,
                      const float* __restrict__ bias, const float* __restrict__ Res,
                      float* __restrict__ C, int M, int N, int K,
                      long long aBatch, long long cBatch) {
    __shared__ __align__(32) __bf16 As[128 * 32];  // 8 KB
    __shared__ __align__(32) __bf16 BsT[64 * 32];  // 4 KB (N-major)

    const int tid = threadIdx.x, lane = tid & 31, wave = tid >> 5;
    const int m0 = blockIdx.y * 128, n0 = blockIdx.x * 64;
    const float* Ab = A + (long long)blockIdx.z * aBatch;
    float* Cb = C + (long long)blockIdx.z * cBatch;
    const float* Rb = RES ? (Res + (long long)blockIdx.z * cBatch) : nullptr;

    v8f acc[4];
#pragma unroll
    for (int t = 0; t < 4; ++t)
#pragma unroll
        for (int i = 0; i < 8; ++i) acc[t][i] = 0.0f;

    for (int k0 = 0; k0 < K; k0 += 32) {
        // prefetch next K-step lines toward the WGP while WMMAs drain this one
        if (k0 + 32 < K) {
            __builtin_prefetch(Ab + (long long)(m0 + (tid >> 1)) * K + k0 + 32, 0, 3);
            __builtin_prefetch(BT + (long long)(n0 + (tid >> 2)) * K + k0 + 32, 0, 3);
        }
        // A tile: 128x32 fp32 -> bf16, vectorized (float4 load + packed b64 LDS store)
#pragma unroll
        for (int i = 0; i < 4; ++i) {
            int e = tid + i * 256;                    // 1024 float4 chunks
            int r = e >> 3, c4 = e & 7;
            float4 av = *(const float4*)(Ab + (long long)(m0 + r) * K + k0 + c4 * 4);
            uint2 u;
            u.x = pack_bf16(av.x, av.y);
            u.y = pack_bf16(av.z, av.w);
            *(uint2*)(As + r * 32 + c4 * 4) = u;
        }
        // B tile: 64 cols x 32 ks from pre-transposed bf16 (one b128 load+store per thread)
        {
            int nn = tid >> 2, kk8 = tid & 3;         // 64 x 4 chunks of 8 halves
            v8bf bv = *(const v8bf*)(BT + (long long)(n0 + nn) * K + k0 + kk8 * 8);
            *(v8bf*)(BsT + nn * 32 + kk8 * 8) = bv;
        }
        __syncthreads();
        v16bf a = frag_a(As, 32, wave * 16, 0, lane);
#pragma unroll
        for (int t = 0; t < 4; ++t) {
            v16bf b = frag_b(BsT, 32, t * 16, 0, lane);
            acc[t] = wmma_bf16(a, b, acc[t]);
        }
        __syncthreads();
    }
    // epilogue: C layout -> VGPR r: M=r (+8 upper half-wave), N = lane&15
    int rbase = m0 + wave * 16 + ((lane & 16) ? 8 : 0);
    int cbase = n0 + (lane & 15);
#pragma unroll
    for (int t = 0; t < 4; ++t) {
        int col = cbase + t * 16;
        float bv = BIAS ? bias[col] : 0.0f;
#pragma unroll
        for (int r = 0; r < 8; ++r) {
            int row = rbase + r;
            float v = acc[t][r] + bv;
            if (ACT == 1) v = gelu_f(v);
            if (RES) v += Rb[(long long)row * N + col];
            Cb[(long long)row * N + col] = v;
        }
    }
}

// ---------------- small elementwise / reduction kernels ----------------
__global__ __launch_bounds__(256)
void tsym_kernel(const float* __restrict__ topo, float* __restrict__ tsym) {
    int idx = blockIdx.x * 256 + threadIdx.x;           // over B*N*N
    int b = idx >> 14, rem = idx & (NN_ - 1);
    int i = rem >> 7, j = rem & (N_ - 1);
    tsym[idx] = topo[idx] + topo[b * NN_ + j * N_ + i];
}

__global__ __launch_bounds__(256)
void add_emb_kernel(float* __restrict__ x, const float* __restrict__ weight,
                    const float* __restrict__ w_w, const float* __restrict__ b_w,
                    const float* __restrict__ n_emb) {
    int idx = blockIdx.x * 256 + threadIdx.x;           // over TN*H
    int row = idx >> 9, hcol = idx & (H_ - 1);
    int n = row & (N_ - 1);
    x[idx] += weight[row] * w_w[hcol] + b_w[hcol] + n_emb[n * H_ + hcol];
}

__global__ __launch_bounds__(256)
void ln_kernel(const float* __restrict__ X, float* __restrict__ Y) {
    __shared__ float red[256];
    int tid = threadIdx.x;
    const float* x = X + (long long)blockIdx.x * H_;
    float s = 0.0f;
    for (int c = tid; c < H_; c += 256) s += x[c];
    red[tid] = s; __syncthreads();
    for (int st = 128; st > 0; st >>= 1) { if (tid < st) red[tid] += red[tid + st]; __syncthreads(); }
    float mean = red[0] / (float)H_; __syncthreads();
    float v = 0.0f;
    for (int c = tid; c < H_; c += 256) { float d = x[c] - mean; v += d * d; }
    red[tid] = v; __syncthreads();
    for (int st = 128; st > 0; st >>= 1) { if (tid < st) red[tid] += red[tid + st]; __syncthreads(); }
    float rstd = rsqrtf(red[0] / (float)H_ + 1e-6f);
    float* y = Y + (long long)blockIdx.x * H_;
    for (int c = tid; c < H_; c += 256) y[c] = (x[c] - mean) * rstd;
}

__global__ __launch_bounds__(256)
void init_att_kernel(float* __restrict__ agg, float* __restrict__ m, float* __restrict__ z) {
    int idx = blockIdx.x * 256 + threadIdx.x;           // over TN*H
    agg[idx] = 0.0f;
    if (idx < TN_ * NH_) { m[idx] = -1e9f; z[idx] = 0.0f; }
}

__global__ __launch_bounds__(256)
void score_kernel(const float* __restrict__ Q, const float* __restrict__ K,
                  const int* __restrict__ ei, int E, float* __restrict__ s) {
    int idx = blockIdx.x * 256 + threadIdx.x;
    if (idx >= E * NH_) return;
    int e = idx >> 3, h = idx & 7;
    int src = ei[e], dst = ei[E + e];
    const float* qv = Q + (long long)dst * H_ + h * HD_;
    const float* kv = K + (long long)src * H_ + h * HD_;
    float acc = 0.0f;
#pragma unroll 8
    for (int k = 0; k < HD_; ++k) acc += qv[k] * kv[k];
    s[idx] = acc * SCALE_;
}

__global__ __launch_bounds__(256)
void segmax_kernel(const float* __restrict__ s, const int* __restrict__ ei, int E,
                   float* __restrict__ m) {
    int idx = blockIdx.x * 256 + threadIdx.x;
    if (idx >= E * NH_) return;
    int e = idx >> 3, h = idx & 7;
    int dst = ei[E + e];
    atomicMaxF(&m[dst * NH_ + h], s[idx]);
}

__global__ __launch_bounds__(256)
void expsum_kernel(const float* __restrict__ s, const int* __restrict__ ei, int E,
                   const float* __restrict__ m, float* __restrict__ ev, float* __restrict__ z) {
    int idx = blockIdx.x * 256 + threadIdx.x;
    if (idx >= E * NH_) return;
    int e = idx >> 3, h = idx & 7;
    int dst = ei[E + e];
    float x = expf(s[idx] - m[dst * NH_ + h]);
    ev[idx] = x;
    atomicAdd(&z[dst * NH_ + h], x);
}

__global__ __launch_bounds__(256)
void aggregate_kernel(const float* __restrict__ ev, const float* __restrict__ z,
                      const float* __restrict__ V, const int* __restrict__ ei, int E,
                      float* __restrict__ agg) {
    int idx = blockIdx.x * 256 + threadIdx.x;
    if (idx >= E * NH_) return;
    int e = idx >> 3, h = idx & 7;
    int src = ei[e], dst = ei[E + e];
    float aw = ev[idx] / (z[dst * NH_ + h] + 1e-12f);
    const float* vv = V + (long long)src * H_ + h * HD_;
    float* av = agg + (long long)dst * H_ + h * HD_;
#pragma unroll 8
    for (int k = 0; k < HD_; ++k) atomicAdd(&av[k], aw * vv[k]);
}

// ---------------- fused all-pairs edge MLP (3-stage, WMMA) ----------------
// 128 pairs per block; f = [xi,xj,xi*xj] (K=1536) built branch-free in 3 phases.
// ew1T: 64x1536 bf16 (pre-transposed), ew2T: 64x64 bf16.
__global__ __launch_bounds__(256)
void edge_mlp_kernel(const float* __restrict__ x,
                     const __bf16* __restrict__ ew1T, const float* __restrict__ eb1,
                     const __bf16* __restrict__ ew2T, const float* __restrict__ eb2,
                     const float* __restrict__ ew3, const float* __restrict__ eb3,
                     float* __restrict__ praw) {
    __shared__ __align__(32) char smem[37888];
    __bf16* As1 = (__bf16*)smem;                 // 128x32  (8 KB)
    __bf16* Bs1 = (__bf16*)(smem + 8192);        // 64x32 T (4 KB)
    __bf16* As2 = (__bf16*)(smem + 12288);       // 128x64  (16 KB)
    __bf16* Bs2 = (__bf16*)(smem + 28672);       // 64x64 T (8 KB)
    float*  g2f = (float*)smem;                  // 128x64 f32 overlay (32 KB)
    float*  red = (float*)(smem + 36864);        // 256 f32

    const int tid = threadIdx.x, lane = tid & 31, wave = tid >> 5;
    const int r0 = blockIdx.x * 128;

    // hoisted per-thread A-staging coordinates (invariant across all K-steps)
    const float* xiP[4];
    const float* xjP[4];
    int kcol[4], ldsOff[4];
#pragma unroll
    for (int i = 0; i < 4; ++i) {
        int e = tid + i * 256;                   // 1024 float4 chunks (128 rows x 8)
        int r = e >> 3, c4 = e & 7;
        int rg = r0 + r;
        int b = rg >> 14, rem = rg & (NN_ - 1);
        int ni = rem >> 7, nj = rem & (N_ - 1);
        xiP[i] = x + (((long long)b * N_ + ni) << 9);
        xjP[i] = x + (((long long)b * N_ + nj) << 9);
        kcol[i]   = c4 * 4;
        ldsOff[i] = r * 32 + c4 * 4;
    }
    const int bnn = tid >> 2, bkk = tid & 3;     // B-staging coords

    v8f acc[4];
#pragma unroll
    for (int t = 0; t < 4; ++t)
#pragma unroll
        for (int i = 0; i < 8; ++i) acc[t][i] = 0.0f;

    // one K-step: stage A (phase-specific, branch-free), stage B, 4 WMMAs
    auto kstep = [&](int kg0, auto pc) {
        constexpr int PH = decltype(pc)::v;
        int kloc = kg0 & 511;                    // offset within x row
#pragma unroll
        for (int i = 0; i < 4; ++i) {
            int kg = kloc + kcol[i];
            float4 av;
            if (PH == 0) {
                av = *(const float4*)(xiP[i] + kg);
            } else if (PH == 1) {
                av = *(const float4*)(xjP[i] + kg);
            } else {
                float4 a1 = *(const float4*)(xiP[i] + kg);
                float4 a2 = *(const float4*)(xjP[i] + kg);
                av.x = a1.x * a2.x; av.y = a1.y * a2.y;
                av.z = a1.z * a2.z; av.w = a1.w * a2.w;
            }
            uint2 u;
            u.x = pack_bf16(av.x, av.y);
            u.y = pack_bf16(av.z, av.w);
            *(uint2*)(As1 + ldsOff[i]) = u;
        }
        {   // ew1T slice: one b128 load+store per thread
            v8bf bv = *(const v8bf*)(ew1T + (long long)bnn * 1536 + kg0 + bkk * 8);
            *(v8bf*)(Bs1 + bnn * 32 + bkk * 8) = bv;
        }
        __syncthreads();
        v16bf a = frag_a(As1, 32, wave * 16, 0, lane);
#pragma unroll
        for (int t = 0; t < 4; ++t) {
            v16bf b = frag_b(Bs1, 32, t * 16, 0, lane);
            acc[t] = wmma_bf16(a, b, acc[t]);
        }
        __syncthreads();
    };
    for (int kc = 0; kc < 16; ++kc) kstep(kc * 32,        IC<0>{});   // f = xi
    for (int kc = 0; kc < 16; ++kc) kstep(512 + kc * 32,  IC<1>{});   // f = xj
    for (int kc = 0; kc < 16; ++kc) kstep(1024 + kc * 32, IC<2>{});   // f = xi*xj

    // g1 = gelu(acc + eb1) -> As2 (bf16 row-major); stage ew2T -> Bs2
    {
        int mlo = wave * 16 + ((lane & 16) ? 8 : 0);
        int cb = lane & 15;
#pragma unroll
        for (int t = 0; t < 4; ++t) {
            int col = t * 16 + cb;
            float bvv = eb1[col];
#pragma unroll
            for (int r = 0; r < 8; ++r)
                As2[(mlo + r) * 64 + col] = (__bf16)gelu_f(acc[t][r] + bvv);
        }
#pragma unroll
        for (int i = 0; i < 2; ++i) {            // 512 chunks of 8 halves
            int e = tid + i * 256;
            int nn = e >> 3, kk8 = e & 7;
            v8bf bv = *(const v8bf*)(ew2T + nn * 64 + kk8 * 8);
            *(v8bf*)(Bs2 + nn * 64 + kk8 * 8) = bv;
        }
    }
    __syncthreads();

    v8f acc2[4];
#pragma unroll
    for (int t = 0; t < 4; ++t)
#pragma unroll
        for (int i = 0; i < 8; ++i) acc2[t][i] = 0.0f;
#pragma unroll
    for (int k0 = 0; k0 < 64; k0 += 32) {
        v16bf a = frag_a(As2, 64, wave * 16, k0, lane);
#pragma unroll
        for (int t = 0; t < 4; ++t) {
            v16bf b = frag_b(Bs2, 64, t * 16, k0, lane);
            acc2[t] = wmma_bf16(a, b, acc2[t]);
        }
    }
    __syncthreads();   // all waves done reading As2/Bs2 before overlay write
    {
        int mlo = wave * 16 + ((lane & 16) ? 8 : 0);
        int cb = lane & 15;
#pragma unroll
        for (int t = 0; t < 4; ++t) {
            int col = t * 16 + cb;
            float bvv = eb2[col];
#pragma unroll
            for (int r = 0; r < 8; ++r)
                g2f[(mlo + r) * 64 + col] = gelu_f(acc2[t][r] + bvv);
        }
    }
    __syncthreads();
    {
        int row = tid & 127, part = tid >> 7;    // 2 threads per row
        float s = 0.0f;
#pragma unroll
        for (int k = 0; k < 32; ++k) {
            int kk = part * 32 + k;
            s += g2f[row * 64 + kk] * ew3[kk];
        }
        red[tid] = s;
    }
    __syncthreads();
    if (tid < 128) {
        float logit = red[tid] + red[tid + 128] + eb3[0];
        praw[r0 + tid] = 1.0f / (1.0f + expf(-logit));
    }
}

__global__ __launch_bounds__(256)
void finalize_kernel(const float* __restrict__ praw, const float* __restrict__ tsym,
                     float* __restrict__ out) {
    int idx = blockIdx.x * 256 + threadIdx.x;    // over B*N*N
    int b = idx >> 14, rem = idx & (NN_ - 1);
    int i = rem >> 7, j = rem & (N_ - 1);
    float v = 0.5f * (praw[idx] + praw[b * NN_ + j * N_ + i]);
    if (i == j || !(tsym[idx] > 0.0f)) v = 0.0f;
    out[idx] = v;
}

// ---------------- launch ----------------
extern "C" void kernel_launch(void* const* d_in, const int* in_sizes, int n_in,
                              void* d_out, int out_size, void* d_ws, size_t ws_size,
                              hipStream_t stream) {
    const float* topo   = (const float*)d_in[0];
    const float* weight = (const float*)d_in[1];
    const int*   ei     = (const int*)d_in[2];
    const float* w_topo = (const float*)d_in[3];
    const float* b_topo = (const float*)d_in[4];
    const float* w_w    = (const float*)d_in[5];
    const float* b_w    = (const float*)d_in[6];
    const float* n_emb  = (const float*)d_in[7];
    const float* wq = (const float*)d_in[8];  const float* bq = (const float*)d_in[9];
    const float* wk = (const float*)d_in[10]; const float* bk = (const float*)d_in[11];
    const float* wv = (const float*)d_in[12]; const float* bv = (const float*)d_in[13];
    const float* wo = (const float*)d_in[14]; const float* bo = (const float*)d_in[15];
    const float* w1 = (const float*)d_in[16]; const float* b1 = (const float*)d_in[17];
    const float* w2 = (const float*)d_in[18]; const float* b2 = (const float*)d_in[19];
    const float* ew1 = (const float*)d_in[20]; const float* eb1 = (const float*)d_in[21];
    const float* ew2 = (const float*)d_in[22]; const float* eb2 = (const float*)d_in[23];
    const float* ew3 = (const float*)d_in[24]; const float* eb3 = (const float*)d_in[25];
    const int E = in_sizes[2] / 2;

    // ---- fp32 workspace ----
    float* p = (float*)d_ws;
    float* tsym = p; p += BNN_;
    float* x    = p; p += TN_ * H_;
    float* h    = p; p += TN_ * H_;
    float* q    = p; p += TN_ * H_;
    float* kk   = p; p += TN_ * H_;
    float* vv   = p; p += TN_ * H_;
    float* agg  = p; p += TN_ * H_;
    float* t1   = p; p += TN_ * MH_;
    float* sbuf = p; p += BNN_ * NH_;
    float* ebuf = p; p += BNN_ * NH_;
    float* mbuf = p; p += TN_ * NH_;
    float* zbuf = p; p += TN_ * NH_;
    float* praw = p; p += BNN_;
    // ---- bf16 (pre-transposed weights) workspace, per-depth buffers reused ----
    __bf16* bp = (__bf16*)p;
    __bf16* wtopoT = bp; bp += H_ * N_;        // 512 x 128
    __bf16* wqT = bp; bp += H_ * H_;
    __bf16* wkT = bp; bp += H_ * H_;
    __bf16* wvT = bp; bp += H_ * H_;
    __bf16* woT = bp; bp += H_ * H_;
    __bf16* w1T = bp; bp += MH_ * H_;          // 2048 x 512
    __bf16* w2T = bp; bp += H_ * MH_;          // 512 x 2048
    __bf16* ew1T = bp; bp += EHD_ * (3 * H_);  // 64 x 1536
    __bf16* ew2T = bp; bp += EHD_ * EHD_;      // 64 x 64

    // 0) one-time per-launch weight converts (shared across layers / tiles)
    convT_kernel<<<dim3((N_ * H_ + 255) / 256, 1, 1), 256, 0, stream>>>(w_topo, wtopoT, N_, H_);
    convT_kernel<<<dim3((3 * H_ * EHD_ + 255) / 256, 1, 1), 256, 0, stream>>>(ew1, ew1T, 3 * H_, EHD_);
    convT_kernel<<<dim3((EHD_ * EHD_ + 255) / 256, 1, 1), 256, 0, stream>>>(ew2, ew2T, EHD_, EHD_);

    // 1) tsym = topo + topo^T
    tsym_kernel<<<BNN_ / 256, 256, 0, stream>>>(topo, tsym);

    // 2) x = tsym @ w_topo + b_topo   (batched over B; per batch M=128,K=128,N=512)
    {
        dim3 g(H_ / 64, N_ / 128, B_);
        gemm_wmma_kernel<0, true, false><<<g, 256, 0, stream>>>(
            tsym, wtopoT, b_topo, nullptr, x, N_, H_, N_,
            (long long)NN_, (long long)(N_ * H_));
    }
    // 3) x += weight_emb + node_emb
    add_emb_kernel<<<(TN_ * H_) / 256, 256, 0, stream>>>(x, weight, w_w, b_w, n_emb);

    const int eh = E * NH_;
    const int gbe = eh > 0 ? (eh + 255) / 256 : 0;
    dim3 gq(H_ / 64, TN_ / 128, 1);
    dim3 g1g(MH_ / 64, TN_ / 128, 1);
    const int gHH = (H_ * H_ + 255) / 256;
    const int gHM = (H_ * MH_ + 255) / 256;

    for (int d = 0; d < DEPTH_; ++d) {
        // per-depth weight converts (buffers reused -> small workspace)
        convT_kernel<<<dim3(gHH, 1, 1), 256, 0, stream>>>(wq + (long long)d * H_ * H_, wqT, H_, H_);
        convT_kernel<<<dim3(gHH, 1, 1), 256, 0, stream>>>(wk + (long long)d * H_ * H_, wkT, H_, H_);
        convT_kernel<<<dim3(gHH, 1, 1), 256, 0, stream>>>(wv + (long long)d * H_ * H_, wvT, H_, H_);
        convT_kernel<<<dim3(gHH, 1, 1), 256, 0, stream>>>(wo + (long long)d * H_ * H_, woT, H_, H_);
        convT_kernel<<<dim3(gHM, 1, 1), 256, 0, stream>>>(w1 + (long long)d * H_ * MH_, w1T, H_, MH_);
        convT_kernel<<<dim3(gHM, 1, 1), 256, 0, stream>>>(w2 + (long long)d * MH_ * H_, w2T, MH_, H_);

        ln_kernel<<<TN_, 256, 0, stream>>>(x, h);
        gemm_wmma_kernel<0, true, false><<<gq, 256, 0, stream>>>(
            h, wqT, bq + d * H_, nullptr, q, TN_, H_, H_, 0, 0);
        gemm_wmma_kernel<0, true, false><<<gq, 256, 0, stream>>>(
            h, wkT, bk + d * H_, nullptr, kk, TN_, H_, H_, 0, 0);
        gemm_wmma_kernel<0, true, false><<<gq, 256, 0, stream>>>(
            h, wvT, bv + d * H_, nullptr, vv, TN_, H_, H_, 0, 0);
        init_att_kernel<<<(TN_ * H_) / 256, 256, 0, stream>>>(agg, mbuf, zbuf);
        if (eh > 0) {
            score_kernel<<<gbe, 256, 0, stream>>>(q, kk, ei, E, sbuf);
            segmax_kernel<<<gbe, 256, 0, stream>>>(sbuf, ei, E, mbuf);
            expsum_kernel<<<gbe, 256, 0, stream>>>(sbuf, ei, E, mbuf, ebuf, zbuf);
            aggregate_kernel<<<gbe, 256, 0, stream>>>(ebuf, zbuf, vv, ei, E, agg);
        }
        // x = agg @ wo + bo + x
        gemm_wmma_kernel<0, true, true><<<gq, 256, 0, stream>>>(
            agg, woT, bo + d * H_, x, x, TN_, H_, H_, 0, 0);
        ln_kernel<<<TN_, 256, 0, stream>>>(x, h);
        // t1 = gelu(h @ w1 + b1)
        gemm_wmma_kernel<1, true, false><<<g1g, 256, 0, stream>>>(
            h, w1T, b1 + d * MH_, nullptr, t1, TN_, MH_, H_, 0, 0);
        // x = t1 @ w2 + b2 + x
        gemm_wmma_kernel<0, true, true><<<gq, 256, 0, stream>>>(
            t1, w2T, b2 + d * H_, x, x, TN_, H_, MH_, 0, 0);
    }

    // edge MLP over all B*N*N pairs, then symmetrize + mask
    edge_mlp_kernel<<<BNN_ / 128, 256, 0, stream>>>(x, ew1T, eb1, ew2T, eb2, ew3, eb3, praw);
    finalize_kernel<<<BNN_ / 256, 256, 0, stream>>>(praw, tsym, (float*)d_out);
}
